// Eagle3LlamaAttention_7310034338444
// MI455X (gfx1250) — matrix operand order repaired and checked by reading, not compiled, not run on hardware
//
#include <hip/hip_runtime.h>
#include <hip/hip_bf16.h>
#include <stdint.h>

// ---------------- problem constants ----------------
#define Bc    2
#define Tc    2048
#define HIDc  2048
#define Hc    32
#define KVc   8
#define Dc    64
#define MT    (Bc*Tc)          // 4096 rows
#define KX    (2*HIDc)         // 4096 input features
#define SCALEc 0.125f          // 64^-0.5
#define STEPSc 2

typedef __bf16 bf16_t;
typedef __attribute__((ext_vector_type(16))) __bf16 v16bf;
typedef __attribute__((ext_vector_type(8)))  __bf16 v8bf;
typedef __attribute__((ext_vector_type(4)))  __bf16 v4bf;
typedef __attribute__((ext_vector_type(8)))  float  v8f;

__device__ __forceinline__ bf16_t f2bf(float f) {
    uint32_t u = __builtin_bit_cast(uint32_t, f);
    uint32_t r = u + 0x7FFFu + ((u >> 16) & 1u);   // RNE
    uint16_t h = (uint16_t)(r >> 16);
    return __builtin_bit_cast(bf16_t, h);
}

__device__ __forceinline__ v16bf combine16(v8bf lo, v8bf hi) {
    v16bf r;
#pragma unroll
    for (int i = 0; i < 8; i++) { r[i] = lo[i]; r[i + 8] = hi[i]; }
    return r;
}

// ---------------- convert f32 -> bf16 (x4 vectorized) ----------------
__global__ void cvt_f32_bf16(const float* __restrict__ src, bf16_t* __restrict__ dst, int n4) {
    int i = blockIdx.x * blockDim.x + threadIdx.x;
    if (i >= n4) return;
    float4 v = ((const float4*)src)[i];
    v4bf o;
    o[0] = f2bf(v.x); o[1] = f2bf(v.y); o[2] = f2bf(v.z); o[3] = f2bf(v.w);
    *(v4bf*)(dst + (size_t)i * 4) = o;
}

// ---------------- convert + transpose weights: W[K][N] f32 -> Wt[N][K] bf16 ----------------
__global__ void tcvt_f32_bf16(const float* __restrict__ W, bf16_t* __restrict__ Wt, int K, int N) {
    __shared__ float tile[32][33];
    int n0 = blockIdx.x * 32, k0 = blockIdx.y * 32;
    int tx = threadIdx.x, ty = threadIdx.y;    // 32 x 8
#pragma unroll
    for (int i = 0; i < 32; i += 8)
        tile[ty + i][tx] = W[(size_t)(k0 + ty + i) * N + n0 + tx];
    __syncthreads();
#pragma unroll
    for (int i = 0; i < 32; i += 8)
        Wt[(size_t)(n0 + ty + i) * K + k0 + tx] = f2bf(tile[tx][ty + i]);
}

// ---------------- bf16 WMMA GEMM: C[M][N] (f32) = A[M][K] (bf16) * Wt[N][K]^T ----------------
// block = 256 threads (8 waves as 2Mx4N); block tile 128x128; wave tile 64x32 (4x2 of 16x16)
// staging via CDNA5 async global->LDS DMA (ASYNCcnt), no VGPR round-trip.
#define ASTR 40
__global__ void gemm_bf16(const bf16_t* __restrict__ A, const bf16_t* __restrict__ Bt,
                          float* __restrict__ C, int M, int K, int N) {
    __shared__ bf16_t As[128 * ASTR];
    __shared__ bf16_t Bs[128 * ASTR];
    const int tid   = threadIdx.x;
    const int wave  = tid >> 5, lane = tid & 31;
    const int upper = lane >> 4, nl = lane & 15;
    const int off   = upper * 8;
    const int wm = wave >> 2, wn = wave & 3;
    const int mbase = blockIdx.y * 128, nbase = blockIdx.x * 128;

    v8f acc[4][2];
#pragma unroll
    for (int i = 0; i < 4; i++)
#pragma unroll
        for (int j = 0; j < 2; j++)
#pragma unroll
            for (int r = 0; r < 8; r++) acc[i][j][r] = 0.f;

    const int srow = tid >> 1, shalf = (tid & 1) * 16;       // staging: 32B per thread per matrix
    const bf16_t* aptr = A  + (size_t)(mbase + srow) * K + shalf;
    const bf16_t* bptr = Bt + (size_t)(nbase + srow) * K + shalf;
    // low 32 bits of the generic (shared-aperture) address = LDS byte offset
    const uint32_t as_lds = (uint32_t)(uintptr_t)(As + srow * ASTR + shalf);
    const uint32_t bs_lds = (uint32_t)(uintptr_t)(Bs + srow * ASTR + shalf);

    for (int k0 = 0; k0 < K; k0 += 32) {
        __syncthreads();                       // everyone done reading previous tile
        uint64_t ga = (uint64_t)(uintptr_t)(aptr + k0);
        uint64_t gb = (uint64_t)(uintptr_t)(bptr + k0);
        // async DMA: 32B of A and 32B of B per thread straight into LDS
        asm volatile(
            "global_load_async_to_lds_b128 %0, %2, off\n\t"
            "global_load_async_to_lds_b128 %0, %2, off offset:16\n\t"
            "global_load_async_to_lds_b128 %1, %3, off\n\t"
            "global_load_async_to_lds_b128 %1, %3, off offset:16"
            :
            : "v"(as_lds), "v"(bs_lds), "v"(ga), "v"(gb)
            : "memory");
        __builtin_prefetch(aptr + k0 + 32, 0, 1);
        __builtin_prefetch(bptr + k0 + 32, 0, 1);
        asm volatile("s_wait_asynccnt 0" ::: "memory");
        __syncthreads();                       // all waves' async data landed

        v16bf af[4], bfr[2];
#pragma unroll
        for (int i = 0; i < 4; i++) {
            const bf16_t* p = As + (wm * 64 + i * 16 + nl) * ASTR;
            af[i] = combine16(*(const v8bf*)(p + off), *(const v8bf*)(p + 16 + off));
        }
#pragma unroll
        for (int j = 0; j < 2; j++) {
            const bf16_t* p = Bs + (wn * 32 + j * 16 + nl) * ASTR;
            bfr[j] = combine16(*(const v8bf*)(p + off), *(const v8bf*)(p + 16 + off));
        }
#pragma unroll
        for (int i = 0; i < 4; i++)
#pragma unroll
            for (int j = 0; j < 2; j++)
                acc[i][j] = __builtin_amdgcn_wmma_f32_16x16x32_bf16(
                    false, af[i], false, bfr[j], (short)0, acc[i][j], false, false);
    }

#pragma unroll
    for (int i = 0; i < 4; i++)
#pragma unroll
        for (int j = 0; j < 2; j++) {
            int row0 = mbase + wm * 64 + i * 16 + upper * 8;
            int col  = nbase + wn * 32 + j * 16 + nl;
            float* cp = C + (size_t)row0 * N + col;
#pragma unroll
            for (int r = 0; r < 8; r++)
                cp[(size_t)r * N] = acc[i][j][r];
        }
}

// ---------------- RoPE in place on x[(B,T,heads,64)] f32, pos = t + STEPS ----------------
__global__ void rope_kernel(float* __restrict__ x, int heads) {
    int i = blockIdx.x * blockDim.x + threadIdx.x;
    int d = i & 31;
    int rest = i >> 5;                       // (b*T + t)*heads + h
    int t = (rest / heads) % Tc;
    float ang = (float)(t + STEPSc) * __expf(-(float)d * (9.210340371976184f / 32.0f));
    float s, c;
    __sincosf(ang, &s, &c);
    size_t base = (size_t)rest * 64;
    float x1 = x[base + d], x2 = x[base + d + 32];
    x[base + d]      = x1 * c - x2 * s;
    x[base + d + 32] = x2 * c + x1 * s;
}

// ---------------- flash attention ----------------
// grid = (T/128, B*H), block = 256 (8 waves, each owns a 16-query tile)
// 32 keys per iteration: 4 score WMMAs + 4 full-K P*V WMMAs
#define PSTR 40
#define VSTR 40
__global__ void attn_kernel(const float*  __restrict__ qf,   // (B,T,H,64)  f32 roped
                            const bf16_t* __restrict__ qb,   // (B,T,H,64)  bf16 roped
                            const float*  __restrict__ kf,   // (B,T,KV,64) f32 roped (k_new)
                            const float*  __restrict__ vf,   // (B,T,KV,64) f32 (v_new)
                            const bf16_t* __restrict__ k0b,  // (B,H,T,64)  bf16  = prev_k[0]
                            const float*  __restrict__ pk1,  // (B,H,T,64)  f32   = prev_k[1]
                            const float*  __restrict__ pv0,  // (B,H,T,64)  f32   = prev_v[0]
                            const float*  __restrict__ pv1,  // (B,H,T,64)  f32   = prev_v[1]
                            bf16_t* __restrict__ ao) {       // (B,T,H*64)  bf16
    __shared__ bf16_t lds[8][16 * PSTR + 64 * VSTR];   // per wave: P 16x32 (pad 40), Vt 64x32 (pad 40)
    const int tid = threadIdx.x, wave = tid >> 5, lane = tid & 31;
    const int upper = lane >> 4, nl = lane & 15, off = upper * 8;
    const int bh = blockIdx.y, b = bh >> 5, h = bh & 31, kvh = h >> 2;
    const int t0 = blockIdx.x * 128 + wave * 16;
    bf16_t* ldsP = &lds[wave][0];
    bf16_t* ldsV = &lds[wave][16 * PSTR];

    // ---- Q fragments (two K=32 chunks of the 16x64 tile) ----
    const bf16_t* qrow = qb + ((size_t)(b * Tc + t0 + nl) * Hc + h) * 64;
    v16bf aq0 = combine16(*(const v8bf*)(qrow + 2 * off),      *(const v8bf*)(qrow + 8 + 2 * off));
    v16bf aq1 = combine16(*(const v8bf*)(qrow + 32 + 2 * off), *(const v8bf*)(qrow + 40 + 2 * off));

    float Mr[8], Lr[8], corr[8];
    v8f O[4];
#pragma unroll
    for (int r = 0; r < 8; r++) { Mr[r] = -1e30f; Lr[r] = 0.f; }
#pragma unroll
    for (int c = 0; c < 4; c++)
#pragma unroll
        for (int r = 0; r < 8; r++) O[c][r] = 0.f;

    const bf16_t* kbase = k0b + ((size_t)(b * Hc + h) * Tc) * 64;
    const float*  vbase = pv0 + ((size_t)(b * Hc + h) * Tc) * 64;

    for (int s0 = 0; s0 <= t0; s0 += 32) {
        // ---- two 16x16 score tiles: keys [s0, s0+16) and [s0+16, s0+32) ----
        const bf16_t* krowA = kbase + (size_t)(s0 + nl) * 64 + upper * 16;
        const bf16_t* krowB = krowA + 16 * 64;
        v16bf bk0a = *(const v16bf*)(krowA);
        v16bf bk1a = *(const v16bf*)(krowA + 32);
        v16bf bk0b = *(const v16bf*)(krowB);
        v16bf bk1b = *(const v16bf*)(krowB + 32);
        v8f sa, sb;
#pragma unroll
        for (int i = 0; i < 8; i++) { sa[i] = 0.f; sb[i] = 0.f; }
        sa = __builtin_amdgcn_wmma_f32_16x16x32_bf16(false, aq0, false, bk0a, (short)0, sa, false, false);
        sa = __builtin_amdgcn_wmma_f32_16x16x32_bf16(false, aq1, false, bk1a, (short)0, sa, false, false);
        sb = __builtin_amdgcn_wmma_f32_16x16x32_bf16(false, aq0, false, bk0b, (short)0, sb, false, false);
        sb = __builtin_amdgcn_wmma_f32_16x16x32_bf16(false, aq1, false, bk1b, (short)0, sb, false, false);

        // ---- online softmax over 32 columns ----
#pragma unroll
        for (int r = 0; r < 8; r++) {
            int mrow = r + upper * 8;
            int trow = t0 + mrow;
            float xa = sa[r] * SCALEc;
            float xb = sb[r] * SCALEc;
            if (s0 + nl > trow)      xa = -1e9f;     // causal
            if (s0 + 16 + nl > trow) xb = -1e9f;
            float rm = fmaxf(xa, xb);
#pragma unroll
            for (int xm = 1; xm < 16; xm <<= 1) rm = fmaxf(rm, __shfl_xor(rm, xm, 16));
            float nM = fmaxf(Mr[r], rm);
            float pa = __expf(xa - nM), pb = __expf(xb - nM);
            float rs = pa + pb;
#pragma unroll
            for (int xm = 1; xm < 16; xm <<= 1) rs += __shfl_xor(rs, xm, 16);
            float cr = __expf(Mr[r] - nM);
            Lr[r] = Lr[r] * cr + rs;
            Mr[r] = nM;
            corr[r] = cr;
            ldsP[mrow * PSTR + nl]      = f2bf(pa);
            ldsP[mrow * PSTR + 16 + nl] = f2bf(pb);
        }

        // ---- stage V0 32x64 tile transposed into LDS as bf16 [d][s] ----
        const float* vt = vbase + (size_t)s0 * 64;
#pragma unroll
        for (int it = 0; it < 16; it++) {
            int flat = (it * 32 + lane) * 4;
            int sv = flat >> 6, d = flat & 63;
            float4 vv = *(const float4*)(vt + sv * 64 + d);
            ldsV[(d + 0) * VSTR + sv] = f2bf(vv.x);
            ldsV[(d + 1) * VSTR + sv] = f2bf(vv.y);
            ldsV[(d + 2) * VSTR + sv] = f2bf(vv.z);
            ldsV[(d + 3) * VSTR + sv] = f2bf(vv.w);
        }
        asm volatile("s_wait_dscnt 0" ::: "memory");   // same-wave LDS store->load fence

        // ---- P fragment (16x32, full K) ----
        const bf16_t* pp = ldsP + nl * PSTR;
        v16bf ap = combine16(*(const v8bf*)(pp + off), *(const v8bf*)(pp + 16 + off));

        // ---- O = O*corr + P * V0 (four 16-wide d chunks, all lanes busy) ----
#pragma unroll
        for (int c = 0; c < 4; c++) {
            const bf16_t* vp = ldsV + (c * 16 + nl) * VSTR;
            v16bf bv = combine16(*(const v8bf*)(vp + off), *(const v8bf*)(vp + 16 + off));
#pragma unroll
            for (int r = 0; r < 8; r++) O[c][r] *= corr[r];
            O[c] = __builtin_amdgcn_wmma_f32_16x16x32_bf16(false, ap, false, bv, (short)0, O[c], false, false);
        }
    }

    // ---- two extra "diagonal" keys: later_k[0]=prev_k[1], later_k[1]=k_new ----
    const float* k1b = pk1 + ((size_t)(b * Hc + h) * Tc) * 64;
    const float* v1b = pv1 + ((size_t)(b * Hc + h) * Tc) * 64;
#pragma unroll
    for (int r = 0; r < 8; r++) {
        int t = t0 + r + upper * 8;
        const float* qr  = qf + ((size_t)(b * Tc + t) * Hc + h) * 64;
        const float* kr1 = k1b + (size_t)t * 64;
        const float* kr2 = kf + ((size_t)(b * Tc + t) * KVc + kvh) * 64;
        float d1 = 0.f, d2 = 0.f;
#pragma unroll
        for (int j = 0; j < 4; j++) {
            float qv = qr[nl * 4 + j];
            d1 += qv * kr1[nl * 4 + j];
            d2 += qv * kr2[nl * 4 + j];
        }
#pragma unroll
        for (int xm = 1; xm < 16; xm <<= 1) {
            d1 += __shfl_xor(d1, xm, 16);
            d2 += __shfl_xor(d2, xm, 16);
        }
        float e1 = d1 * SCALEc, e2 = d2 * SCALEc;
        float nM = fmaxf(Mr[r], fmaxf(e1, e2));
        float cr = __expf(Mr[r] - nM);
        float p1 = __expf(e1 - nM), p2 = __expf(e2 - nM);
        Lr[r] = Lr[r] * cr + p1 + p2;
        Mr[r] = nM;
        const float* vr1 = v1b + (size_t)t * 64;
        const float* vr2 = vf + ((size_t)(b * Tc + t) * KVc + kvh) * 64;
#pragma unroll
        for (int c = 0; c < 4; c++)
            O[c][r] = O[c][r] * cr + p1 * vr1[c * 16 + nl] + p2 * vr2[c * 16 + nl];
    }

    // ---- normalize and write (B,T,H*64) bf16 ----
#pragma unroll
    for (int r = 0; r < 8; r++) {
        int t = t0 + r + upper * 8;
        float inv = 1.0f / Lr[r];
        bf16_t* orow = ao + (size_t)(b * Tc + t) * (Hc * Dc) + h * 64;
#pragma unroll
        for (int c = 0; c < 4; c++)
            orow[c * 16 + nl] = f2bf(O[c][r] * inv);
    }
}

// ---------------- host side ----------------
extern "C" void kernel_launch(void* const* d_in, const int* in_sizes, int n_in,
                              void* d_out, int out_size, void* d_ws, size_t ws_size,
                              hipStream_t stream) {
    const float* X  = (const float*)d_in[0];   // (B,T,4096)
    const float* pk = (const float*)d_in[3];   // (2,B,H,T,64)
    const float* pv = (const float*)d_in[4];
    const float* Wq = (const float*)d_in[5];   // (4096,2048)
    const float* Wk = (const float*)d_in[6];   // (4096,512)
    const float* Wv = (const float*)d_in[7];   // (4096,512)
    const float* Wo = (const float*)d_in[8];   // (2048,2048)
    float* out = (float*)d_out;                // (B,T,2048) f32

    const size_t KV_STEP = (size_t)Bc * Hc * Tc * Dc;   // 8,388,608 elems per step slice

    char* ws = (char*)d_ws;
    size_t o = 0;
    auto alloc = [&](size_t bytes) { void* p = ws + o; o += (bytes + 255) & ~(size_t)255; return p; };

    bf16_t* Xb  = (bf16_t*)alloc((size_t)MT * KX * 2);          // combined bf16
    bf16_t* Wqt = (bf16_t*)alloc((size_t)2048 * KX * 2);        // [N][K]
    bf16_t* Wkt = (bf16_t*)alloc((size_t)512 * KX * 2);
    bf16_t* Wvt = (bf16_t*)alloc((size_t)512 * KX * 2);
    bf16_t* Wot = (bf16_t*)alloc((size_t)2048 * 2048 * 2);
    float*  qfb = (float*)alloc((size_t)MT * 2048 * 4);         // (B,T,H,64)
    float*  kfb = (float*)alloc((size_t)MT * 512 * 4);          // (B,T,KV,64)
    float*  vfb = (float*)alloc((size_t)MT * 512 * 4);
    bf16_t* qbb = (bf16_t*)alloc((size_t)MT * 2048 * 2);
    bf16_t* k0b = (bf16_t*)alloc(KV_STEP * 2);
    bf16_t* aob = (bf16_t*)alloc((size_t)MT * 2048 * 2);

    // 1. convert activations + weights to bf16 (weights transposed for WMMA B layout)
    cvt_f32_bf16<<<(MT * KX / 4) / 256, 256, 0, stream>>>(X, Xb, MT * KX / 4);
    tcvt_f32_bf16<<<dim3(2048 / 32, KX / 32), dim3(32, 8), 0, stream>>>(Wq, Wqt, KX, 2048);
    tcvt_f32_bf16<<<dim3(512 / 32,  KX / 32), dim3(32, 8), 0, stream>>>(Wk, Wkt, KX, 512);
    tcvt_f32_bf16<<<dim3(512 / 32,  KX / 32), dim3(32, 8), 0, stream>>>(Wv, Wvt, KX, 512);
    tcvt_f32_bf16<<<dim3(2048 / 32, 2048 / 32), dim3(32, 8), 0, stream>>>(Wo, Wot, 2048, 2048);

    // 2. QKV projections (WMMA, 128x128 block tiles, async-LDS staging)
    gemm_bf16<<<dim3(2048 / 128, MT / 128), 256, 0, stream>>>(Xb, Wqt, qfb, MT, KX, 2048);
    gemm_bf16<<<dim3(512 / 128,  MT / 128), 256, 0, stream>>>(Xb, Wkt, kfb, MT, KX, 512);
    gemm_bf16<<<dim3(512 / 128,  MT / 128), 256, 0, stream>>>(Xb, Wvt, vfb, MT, KX, 512);

    // 3. RoPE (pos = t + STEPS)
    rope_kernel<<<(MT * Hc * 32) / 256, 256, 0, stream>>>(qfb, Hc);
    rope_kernel<<<(MT * KVc * 32) / 256, 256, 0, stream>>>(kfb, KVc);

    // 4. bf16 copies for the attention WMMAs
    cvt_f32_bf16<<<(MT * 2048 / 4) / 256, 256, 0, stream>>>(qfb, qbb, MT * 2048 / 4);
    cvt_f32_bf16<<<((int)KV_STEP / 4) / 256, 256, 0, stream>>>(pk, k0b, (int)KV_STEP / 4);

    // 5. flash attention (WMMA scores + WMMA P*V0 + 2 diagonal keys)
    attn_kernel<<<dim3(Tc / 128, Bc * Hc), 256, 0, stream>>>(
        qfb, qbb, kfb, vfb, k0b,
        pk + KV_STEP,            // prev_k[1]
        pv,                      // prev_v[0]
        pv + KV_STEP,            // prev_v[1]
        aob);

    // 6. output projection (WMMA) -> f32 d_out
    gemm_bf16<<<dim3(2048 / 128, MT / 128), 256, 0, stream>>>(aob, Wot, out, MT, 2048, 2048);
}
